// NodeInfoPropagator_38525856645684
// MI455X (gfx1250) — matrix operand order repaired and checked by reading, not compiled, not run on hardware
//
#include <hip/hip_runtime.h>
#include <hip/hip_bf16.h>

// ---------------------------------------------------------------------------
// Problem constants (from reference setup_inputs)
// ---------------------------------------------------------------------------
#define BATCH 16
#define NNODE 4096
#define MROWS (BATCH * NNODE)   // 65536
#define PDIM  256               // E == P == 256
#define CDIM  8
#define DEPTH 3
#define KT256 8                 // 256 / 32 k-tiles

typedef __attribute__((ext_vector_type(16))) __bf16 v16bf;
typedef __attribute__((ext_vector_type(8)))  __bf16 v8bf;
typedef __attribute__((ext_vector_type(8)))  float  v8f;

#define WMMA_BF16(a, b, c) \
    __builtin_amdgcn_wmma_f32_16x16x32_bf16(false, (a), false, (b), (short)0, (c), false, false)

__device__ __forceinline__ float fast_sigmoid(float x) {
    return 1.0f / (1.0f + __expf(-x));
}

// ---------------------------------------------------------------------------
// A-fragment load from LDS (16 rows x 256 bf16, row-major).
// ISA 16-bit A layout (16x32): lane<16 -> row=lane, K {kb+0..7, kb+16..23};
// lane>=16 -> row=lane-16, K {kb+8..15, kb+24..31}.
// ---------------------------------------------------------------------------
__device__ __forceinline__ v16bf a_frag_lds(const __bf16* rows, int kbase, int lane) {
    const int row  = lane & 15;
    const int koff = (lane & 16) ? 8 : 0;
    const __bf16* p = rows + row * PDIM + kbase + koff;
    v8bf lo = *(const v8bf*)p;          // ds_load_b128
    v8bf hi = *(const v8bf*)(p + 16);   // ds_load_b128
    v16bf f;
#pragma unroll
    for (int i = 0; i < 8; ++i) { f[i] = lo[i]; f[8 + i] = hi[i]; }
    return f;
}

// B-fragment load from pre-packed global (L2-resident) weights:
// fragment = [32 lanes][16 bf16] contiguous (1 KB per fragment).
__device__ __forceinline__ v16bf b_frag(const __bf16* pack, int jt, int kt, int ktiles, int lane) {
    return *(const v16bf*)(pack + ((size_t)(jt * ktiles + kt) * 512) + lane * 16);
}

// ---------------------------------------------------------------------------
// Pack fp32 weight W [Nout, K] (row-major) into bf16 WMMA B-fragments.
// B[k][n] = W[n][k]; lane n<16 holds col jt*16+n, K kt*32+{0..7,16..23};
// lane n+16 holds same col, K kt*32+{8..15,24..31}. One wave per fragment.
// ---------------------------------------------------------------------------
__global__ void pack_w_kernel(const float* __restrict__ W, __bf16* __restrict__ out, int K) {
    const int f    = blockIdx.x;
    const int lane = threadIdx.x;            // 0..31
    const int ktiles = K >> 5;
    const int kt = f % ktiles;
    const int jt = f / ktiles;
    const int n  = jt * 16 + (lane & 15);
    const int kb = kt * 32 + ((lane & 16) ? 8 : 0);
    const float* src = W + (size_t)n * K + kb;
    __bf16* dst = out + (size_t)f * 512 + lane * 16;
#pragma unroll
    for (int e = 0; e < 16; ++e) dst[e] = (__bf16)src[e + ((e & 8) ? 8 : 0)];
}

// ---------------------------------------------------------------------------
// h = bf16( x @ W_resize^T + b_resize )   [M,256] x [256,256]
// One workgroup: 16 rows x 256 cols, 16 waves (wave = col tile).
// ---------------------------------------------------------------------------
__global__ __launch_bounds__(512) void resize_kernel(
    const float* __restrict__ x, const __bf16* __restrict__ Wr_pack,
    const float* __restrict__ b_resize, __bf16* __restrict__ h16) {
    __shared__ __bf16 xA[16 * PDIM];
    const int rowbase = blockIdx.x * 16;
    const int tid = threadIdx.x;
    {   // stage + fp32->bf16 convert: thread (i = tid>>5, c = tid&31) does 8 elems
        const int i = tid >> 5, c = tid & 31;
        const float* src = x + (size_t)(rowbase + i) * PDIM + c * 8;
        v8bf o;
#pragma unroll
        for (int e = 0; e < 8; ++e) o[e] = (__bf16)src[e];
        *(v8bf*)(xA + i * PDIM + c * 8) = o;
    }
    __syncthreads();
    const int wave = tid >> 5, lane = tid & 31;
    v8f acc = {};
#pragma unroll
    for (int kt = 0; kt < KT256; ++kt) {
        v16bf a = a_frag_lds(xA, kt * 32, lane);
        v16bf b = b_frag(Wr_pack, wave, kt, KT256, lane);
        acc = WMMA_BF16(a, b, acc);
    }
    const int col = wave * 16 + (lane & 15);
    const float bias = b_resize[col];
    const int m0 = rowbase + ((lane & 16) ? 8 : 0);
#pragma unroll
    for (int r = 0; r < 8; ++r)
        h16[(size_t)(m0 + r) * PDIM + col] = (__bf16)(acc[r] + bias);
}

// ---------------------------------------------------------------------------
// summary = bf16( gather_parent(h)@Wp^T + mean_children(h)@Wn^T + bp + bn )
// Fuses the gathers (L2-resident h rows) into LDS staging.
// ---------------------------------------------------------------------------
__global__ __launch_bounds__(512) void summary_kernel(
    const __bf16* __restrict__ h16,
    const int* __restrict__ parent_idx,
    const int* __restrict__ children_idx,
    const int* __restrict__ child_count,
    const __bf16* __restrict__ Wp_pack,
    const __bf16* __restrict__ Wn_pack,
    const float* __restrict__ b_parent,
    const float* __restrict__ b_nbr,
    __bf16* __restrict__ sum16) {
    __shared__ __bf16 parA[16 * PDIM];
    __shared__ __bf16 nbrA[16 * PDIM];
    const int rowbase = blockIdx.x * 16;
    const int tid = threadIdx.x;
    {   // gather stage: thread (i, c) owns an 8-element chunk of node rowbase+i
        const int i = tid >> 5, c = tid & 31;
        const int m = rowbase + i;
        const int brow = (m >> 12) << 12;         // batch base row (N = 4096)
        const int p = parent_idx[m];
        *(v8bf*)(parA + i * PDIM + c * 8) =
            *((const v8bf*)(h16 + (size_t)(brow + p) * PDIM) + c);
        const int cnt = child_count[m];
        float acc[8];
#pragma unroll
        for (int e = 0; e < 8; ++e) acc[e] = 0.0f;
#pragma unroll
        for (int ch = 0; ch < CDIM; ++ch) {
            if (ch < cnt) {
                const int ci = children_idx[(size_t)m * CDIM + ch];
                v8bf v = *((const v8bf*)(h16 + (size_t)(brow + ci) * PDIM) + c);
#pragma unroll
                for (int e = 0; e < 8; ++e) acc[e] += (float)v[e];
            }
        }
        const float inv = 1.0f / (float)cnt;
        v8bf o;
#pragma unroll
        for (int e = 0; e < 8; ++e) o[e] = (__bf16)(acc[e] * inv);
        *(v8bf*)(nbrA + i * PDIM + c * 8) = o;
    }
    __syncthreads();
    const int wave = tid >> 5, lane = tid & 31;
    v8f acc = {};
#pragma unroll
    for (int kt = 0; kt < KT256; ++kt) {
        v16bf a = a_frag_lds(parA, kt * 32, lane);
        v16bf b = b_frag(Wp_pack, wave, kt, KT256, lane);
        acc = WMMA_BF16(a, b, acc);
    }
#pragma unroll
    for (int kt = 0; kt < KT256; ++kt) {
        v16bf a = a_frag_lds(nbrA, kt * 32, lane);
        v16bf b = b_frag(Wn_pack, wave, kt, KT256, lane);
        acc = WMMA_BF16(a, b, acc);
    }
    const int col = wave * 16 + (lane & 15);
    const float bias = b_parent[col] + b_nbr[col];
    const int m0 = rowbase + ((lane & 16) ? 8 : 0);
#pragma unroll
    for (int r = 0; r < 8; ++r)
        sum16[(size_t)(m0 + r) * PDIM + col] = (__bf16)(acc[r] + bias);
}

// ---------------------------------------------------------------------------
// GRUCell: gx = h@Wih^T + b_ih ; gh = s@Whh^T + b_hh (s = summary = hidden)
// r = sig(xr+hr); z = sig(xz+hz); n = tanh(xn + r*hn); h' = (1-z)*n + z*s
// Wave w owns col tiles {w, w+16, w+32} of the 768-wide gates (r/z/n share a
// P-column), so the elementwise math closes within the wave. 48 WMMAs/wave.
// ---------------------------------------------------------------------------
__global__ __launch_bounds__(512) void gru_kernel(
    const __bf16* __restrict__ h16,
    const __bf16* __restrict__ sum16,
    const __bf16* __restrict__ Wih_pack,   // 48 col tiles x 8 k tiles
    const __bf16* __restrict__ Whh_pack,
    const float* __restrict__ b_ih,
    const float* __restrict__ b_hh,
    __bf16* __restrict__ h16_out,
    float* __restrict__ out32, int write_out) {
    __shared__ __bf16 hA[16 * PDIM];
    __shared__ __bf16 sA[16 * PDIM];
    const int rowbase = blockIdx.x * 16;
    const int tid = threadIdx.x;
    {
        const int i = tid >> 5, c = tid & 31;
        *(v8bf*)(hA + i * PDIM + c * 8) =
            *((const v8bf*)(h16 + (size_t)(rowbase + i) * PDIM) + c);
        *(v8bf*)(sA + i * PDIM + c * 8) =
            *((const v8bf*)(sum16 + (size_t)(rowbase + i) * PDIM) + c);
    }
    __syncthreads();
    const int w = tid >> 5, lane = tid & 31;
    v8f xr = {}, xz = {}, xn = {}, hr = {}, hz = {}, hn = {};
#pragma unroll
    for (int kt = 0; kt < KT256; ++kt) {
        v16bf ah = a_frag_lds(hA, kt * 32, lane);
        v16bf as = a_frag_lds(sA, kt * 32, lane);
        xr = WMMA_BF16(ah, b_frag(Wih_pack, w,      kt, KT256, lane), xr);
        xz = WMMA_BF16(ah, b_frag(Wih_pack, w + 16, kt, KT256, lane), xz);
        xn = WMMA_BF16(ah, b_frag(Wih_pack, w + 32, kt, KT256, lane), xn);
        hr = WMMA_BF16(as, b_frag(Whh_pack, w,      kt, KT256, lane), hr);
        hz = WMMA_BF16(as, b_frag(Whh_pack, w + 16, kt, KT256, lane), hz);
        hn = WMMA_BF16(as, b_frag(Whh_pack, w + 32, kt, KT256, lane), hn);
    }
    const int col = w * 16 + (lane & 15);
    const float bxr = b_ih[col], bxz = b_ih[256 + col], bxn = b_ih[512 + col];
    const float bhr = b_hh[col], bhz = b_hh[256 + col], bhn = b_hh[512 + col];
    const int i0 = (lane & 16) ? 8 : 0;
#pragma unroll
    for (int r = 0; r < 8; ++r) {
        const float rg = fast_sigmoid(xr[r] + bxr + hr[r] + bhr);
        const float zg = fast_sigmoid(xz[r] + bxz + hz[r] + bhz);
        const float ng = tanhf(xn[r] + bxn + rg * (hn[r] + bhn));
        const float s_old = (float)sA[(i0 + r) * PDIM + col];   // hidden == summary
        const float hnew = (1.0f - zg) * ng + zg * s_old;
        const size_t idx = (size_t)(rowbase + i0 + r) * PDIM + col;
        h16_out[idx] = (__bf16)hnew;
        if (write_out) out32[idx] = hnew;
    }
}

// ---------------------------------------------------------------------------
extern "C" void kernel_launch(void* const* d_in, const int* in_sizes, int n_in,
                              void* d_out, int out_size, void* d_ws, size_t ws_size,
                              hipStream_t stream) {
    const float* x            = (const float*)d_in[0];
    const int*   parent_idx   = (const int*)d_in[1];
    const int*   children_idx = (const int*)d_in[2];
    const int*   child_count  = (const int*)d_in[3];
    const float* W_resize     = (const float*)d_in[4];
    const float* b_resize     = (const float*)d_in[5];
    const float* W_parent     = (const float*)d_in[6];
    const float* b_parent     = (const float*)d_in[7];
    const float* W_nbr        = (const float*)d_in[8];
    const float* b_nbr        = (const float*)d_in[9];
    const float* W_ih         = (const float*)d_in[10];
    const float* W_hh         = (const float*)d_in[11];
    const float* b_ih         = (const float*)d_in[12];
    const float* b_hh         = (const float*)d_in[13];
    float* out32 = (float*)d_out;

    // Workspace carve-out (all buffers rebuilt every call; deterministic).
    char* ws = (char*)d_ws;
    size_t off = 0;
    auto carve = [&](size_t bytes) -> void* {
        off = (off + 255) & ~(size_t)255;
        void* p = ws + off;
        off += bytes;
        return p;
    };
    const size_t hbytes = (size_t)MROWS * PDIM * sizeof(__bf16);   // 32 MB
    __bf16* h_a   = (__bf16*)carve(hbytes);
    __bf16* h_b   = (__bf16*)carve(hbytes);
    __bf16* s16   = (__bf16*)carve(hbytes);
    __bf16* WrP   = (__bf16*)carve((size_t)PDIM * PDIM * 2);
    __bf16* WpP   = (__bf16*)carve((size_t)PDIM * PDIM * 2);
    __bf16* WnP   = (__bf16*)carve((size_t)PDIM * PDIM * 2);
    __bf16* WihP  = (__bf16*)carve((size_t)3 * PDIM * PDIM * 2);
    __bf16* WhhP  = (__bf16*)carve((size_t)3 * PDIM * PDIM * 2);
    (void)in_sizes; (void)n_in; (void)out_size; (void)ws_size;

    // 1) pack weights (fragments: [Nout/16]*[K/32], one wave each)
    pack_w_kernel<<<(PDIM / 16) * (PDIM / 32), 32, 0, stream>>>(W_resize, WrP, PDIM);
    pack_w_kernel<<<(PDIM / 16) * (PDIM / 32), 32, 0, stream>>>(W_parent, WpP, PDIM);
    pack_w_kernel<<<(PDIM / 16) * (PDIM / 32), 32, 0, stream>>>(W_nbr,    WnP, PDIM);
    pack_w_kernel<<<(3 * PDIM / 16) * (PDIM / 32), 32, 0, stream>>>(W_ih, WihP, PDIM);
    pack_w_kernel<<<(3 * PDIM / 16) * (PDIM / 32), 32, 0, stream>>>(W_hh, WhhP, PDIM);

    const int nblocks = MROWS / 16;   // 4096

    // 2) h0 = x @ Wr^T + b
    resize_kernel<<<nblocks, 512, 0, stream>>>(x, WrP, b_resize, h_a);

    // 3) DEPTH iterations of gather+summary GEMM then GRU GEMMs
    __bf16* cur = h_a;
    __bf16* nxt = h_b;
    for (int d = 0; d < DEPTH; ++d) {
        summary_kernel<<<nblocks, 512, 0, stream>>>(
            cur, parent_idx, children_idx, child_count,
            WpP, WnP, b_parent, b_nbr, s16);
        gru_kernel<<<nblocks, 512, 0, stream>>>(
            cur, s16, WihP, WhhP, b_ih, b_hh, nxt, out32, (d == DEPTH - 1) ? 1 : 0);
        __bf16* t = cur; cur = nxt; nxt = t;
    }
}
